// StreamFunctionConstraint_80436147519576
// MI455X (gfx1250) — compile-verified, matrix operand order checked by reading, not run on hardware
//
#include <hip/hip_runtime.h>
#include <hip/hip_bf16.h>

typedef __attribute__((ext_vector_type(16))) __bf16 v16bf;
typedef __attribute__((ext_vector_type(8)))  float  v8f;

#define HH 256
#define WW 256
#define CC 64
#define KPAD 72   // channel dim padded 64->72: 144B row stride, 16B aligned, bank-conflict-free

__device__ inline v16bf load_frag16(const __bf16* p) {
  // 32 contiguous bytes, 16B aligned by construction -> ds_load_b128 x2
  union { uint4 q[2]; v16bf v; } u;
  const uint4* s = (const uint4*)p;
  u.q[0] = s[0];
  u.q[1] = s[1];
  return u.v;
}

// ---------------- Kernel 1: h = silu(conv1(z_real)) via bf16 WMMA ----------------
// grid (2, 256, 4), block 256 (8 waves). Each wave: 16 px x 64 oc tile.
__global__ void conv1_silu_wmma(const float* __restrict__ zr,
                                const float* __restrict__ w1,
                                const float* __restrict__ b1,
                                float* __restrict__ h) {
  extern __shared__ char smem[];
  __bf16* tileA = (__bf16*)smem;                          // [3][130][KPAD]
  __bf16* wB    = (__bf16*)(smem + 3 * 130 * KPAD * 2);   // [9][64][KPAD] (tap, oc, cin)

  const int tid = threadIdx.x;
  const int x0  = blockIdx.x * 128;
  const int y   = blockIdx.y;
  const int b   = blockIdx.z;

  // stage all w1 (64x64x3x3) as bf16, layout [tap][oc][cin]
  for (int idx = tid; idx < 9 * 64 * 64; idx += 256) {
    int t   = idx % 9;
    int cin = (idx / 9) % 64;
    int oc  = idx / (9 * 64);
    wB[(t * 64 + oc) * KPAD + cin] = (__bf16)w1[idx];
  }
  // stage input tile: rows y-1..y+1, x0-1..x0+128, all 64 channels (zero pad = SAME)
  for (int idx = tid; idx < 3 * 64 * 130; idx += 256) {
    int x = idx % 130;
    int c = (idx / 130) % 64;
    int r = idx / (130 * 64);
    int gy = y + r - 1;
    int gx = x0 + x - 1;
    float v = 0.f;
    if ((unsigned)gy < (unsigned)HH && (unsigned)gx < (unsigned)WW)
      v = zr[((b * CC + c) * HH + gy) * WW + gx];
    tileA[(r * 130 + x) * KPAD + c] = (__bf16)v;
  }
  __syncthreads();

  const int wv   = tid >> 5;         // wave id 0..7 -> 16-px sub-segment
  const int lane = tid & 31;
  const int mrow = lane & 15;        // A: row M / B: col N / D: col N
  const int klo  = (lane & 16) ? 16 : 0;

  v8f acc[4] = {v8f{}, v8f{}, v8f{}, v8f{}};

#pragma unroll
  for (int t = 0; t < 9; ++t) {
    const int r  = t / 3;            // input row = y + r - 1  -> tileA row r
    const int dx = t % 3;            // input x   = x + dx - 1 -> tile idx wv*16+m+dx
#pragma unroll
    for (int kk = 0; kk < 2; ++kk) { // cin chunk of 32
      const __bf16* pa =
          tileA + (r * 130 + (wv * 16 + mrow + dx)) * KPAD + kk * 32 + klo;
      v16bf a = load_frag16(pa);
      const __bf16* pb = wB + (t * 64 + mrow) * KPAD + kk * 32 + klo;
#pragma unroll
      for (int nt = 0; nt < 4; ++nt) {
        v16bf bm = load_frag16(pb + nt * 16 * KPAD);
        acc[nt] = __builtin_amdgcn_wmma_f32_16x16x32_bf16(
            false, a, false, bm, (short)0, acc[nt], false, false);
      }
    }
  }

  // D layout: lane -> N(oc), VGPR r -> M(pixel), lane-half -> +8 pixels
  const int xbase = x0 + wv * 16 + ((lane & 16) ? 8 : 0);
#pragma unroll
  for (int nt = 0; nt < 4; ++nt) {
    const int oc = nt * 16 + mrow;
    const float bias = b1[oc];
    float* dst = h + (((size_t)b * CC + oc) * HH + y) * WW + xbase;
    float4 lo, hi;
    {
      float v0 = acc[nt][0] + bias, v1 = acc[nt][1] + bias,
            v2 = acc[nt][2] + bias, v3 = acc[nt][3] + bias;
      lo.x = v0 / (1.f + __expf(-v0)); lo.y = v1 / (1.f + __expf(-v1));
      lo.z = v2 / (1.f + __expf(-v2)); lo.w = v3 / (1.f + __expf(-v3));
      float v4 = acc[nt][4] + bias, v5 = acc[nt][5] + bias,
            v6 = acc[nt][6] + bias, v7 = acc[nt][7] + bias;
      hi.x = v4 / (1.f + __expf(-v4)); hi.y = v5 / (1.f + __expf(-v5));
      hi.z = v6 / (1.f + __expf(-v6)); hi.w = v7 / (1.f + __expf(-v7));
    }
    *(float4*)dst       = lo;
    *(float4*)(dst + 4) = hi;
  }
}

// ---------------- Kernel 2: psi = (conv2(h) + b2) * dt ----------------
// grid (16,16,4), block 256 -> 16x16 output tile, h tile staged in LDS (L2-resident h)
__global__ void conv2_psi(const float* __restrict__ h,
                          const float* __restrict__ w2,
                          const float* __restrict__ b2,
                          const float* __restrict__ dt,
                          float* __restrict__ psi) {
  extern __shared__ char smem[];
  float* ht  = (float*)smem;          // [64][18][18]
  float* w2s = ht + 64 * 18 * 18;     // [576]

  const int tid = threadIdx.x;
  const int tx = tid & 15, ty = tid >> 4;
  const int bx = blockIdx.x * 16, by = blockIdx.y * 16;
  const int b  = blockIdx.z;

  for (int idx = tid; idx < 576; idx += 256) w2s[idx] = w2[idx];
  for (int idx = tid; idx < 64 * 18 * 18; idx += 256) {
    int xx = idx % 18;
    int yy = (idx / 18) % 18;
    int c  = idx / (18 * 18);
    int gy = by + yy - 1, gx = bx + xx - 1;
    float v = 0.f;
    if ((unsigned)gy < (unsigned)HH && (unsigned)gx < (unsigned)WW)
      v = h[((b * CC + c) * HH + gy) * WW + gx];
    ht[(c * 18 + yy) * 18 + xx] = v;
  }
  __syncthreads();

  float acc = 0.f;
#pragma unroll 4
  for (int c = 0; c < 64; ++c) {
    const float* hp = ht + (c * 18 + ty) * 18 + tx;
    const float* wp = w2s + c * 9;
#pragma unroll
    for (int ky = 0; ky < 3; ++ky)
#pragma unroll
      for (int kx = 0; kx < 3; ++kx)
        acc = fmaf(hp[ky * 18 + kx], wp[ky * 3 + kx], acc);
  }
  psi[((size_t)b * HH + (by + ty)) * WW + (bx + tx)] = (acc + b2[0]) * dt[0];
}

// ---------------- Kernel 3: curl(psi) -> flow -> bilinear border sample ----------------
__global__ void curl_sample(const float* __restrict__ psi,
                            const float* __restrict__ zr,
                            const float* __restrict__ zi,
                            float* __restrict__ out) {
  const int gid = blockIdx.x * 256 + threadIdx.x;  // b*H*W threads
  const int x = gid & (WW - 1);
  const int y = (gid >> 8) & (HH - 1);
  const int b = gid >> 16;
  const float* p = psi + (size_t)b * HH * WW;

  const int ym = max(y - 1, 0), yp = min(y + 1, HH - 1);
  const int xm = max(x - 1, 0), xp = min(x + 1, WW - 1);
  const float u =  0.5f * (p[yp * WW + x] - p[ym * WW + x]);   // d psi / dy
  const float v = -0.5f * (p[y * WW + xp] - p[y * WW + xm]);   // -d psi / dx
  const float flow_x = u * (2.f / WW);
  const float flow_y = v * (2.f / HH);

  const float basex = -1.f + 2.f * (float)x / (float)(WW - 1);
  const float basey = -1.f + 2.f * (float)y / (float)(HH - 1);
  const float gx = basex - flow_x;
  const float gy = basey - flow_y;

  float ix = fminf(fmaxf((gx + 1.f) * 0.5f * (float)(WW - 1), 0.f), (float)(WW - 1));
  float iy = fminf(fmaxf((gy + 1.f) * 0.5f * (float)(HH - 1), 0.f), (float)(HH - 1));
  const float x0f = floorf(ix), y0f = floorf(iy);
  const int x0 = (int)x0f, y0 = (int)y0f;
  const int x1 = min(x0 + 1, WW - 1), y1 = min(y0 + 1, HH - 1);
  const float wx = ix - x0f, wy = iy - y0f;
  const float w00 = (1.f - wx) * (1.f - wy), w01 = wx * (1.f - wy);
  const float w10 = (1.f - wx) * wy,         w11 = wx * wy;
  const int i00 = y0 * WW + x0, i01 = y0 * WW + x1;
  const int i10 = y1 * WW + x0, i11 = y1 * WW + x1;
  const int opix = y * WW + x;

#pragma unroll 4
  for (int c = 0; c < CC; ++c) {
    const float* img = zr + ((size_t)b * CC + c) * HH * WW;
    out[((size_t)b * 2 * CC + c) * HH * WW + opix] =
        img[i00] * w00 + img[i01] * w01 + img[i10] * w10 + img[i11] * w11;
  }
#pragma unroll 4
  for (int c = 0; c < CC; ++c) {
    const float* img = zi + ((size_t)b * CC + c) * HH * WW;
    out[((size_t)b * 2 * CC + (CC + c)) * HH * WW + opix] =
        img[i00] * w00 + img[i01] * w01 + img[i10] * w10 + img[i11] * w11;
  }
}

extern "C" void kernel_launch(void* const* d_in, const int* in_sizes, int n_in,
                              void* d_out, int out_size, void* d_ws, size_t ws_size,
                              hipStream_t stream) {
  const float* zr = (const float*)d_in[0];  // z_real [4,64,256,256]
  const float* zi = (const float*)d_in[1];  // z_imag
  const float* dt = (const float*)d_in[2];  // scalar
  const float* w1 = (const float*)d_in[3];  // [64,64,3,3]
  const float* b1 = (const float*)d_in[4];  // [64]
  const float* w2 = (const float*)d_in[5];  // [1,64,3,3]
  const float* b2 = (const float*)d_in[6];  // [1]
  float* out = (float*)d_out;

  float* h   = (float*)d_ws;                         // 4*64*256*256 f32 = 67.1 MB
  float* psi = h + (size_t)4 * CC * HH * WW;         // 4*256*256 f32 = 1 MB

  const size_t lds1 = (size_t)(3 * 130 * KPAD + 9 * 64 * KPAD) * 2;  // ~139 KB
  conv1_silu_wmma<<<dim3(2, HH, 4), 256, lds1, stream>>>(zr, w1, b1, h);

  const size_t lds2 = (size_t)(64 * 18 * 18 + 576) * 4;              // ~85 KB
  conv2_psi<<<dim3(16, 16, 4), 256, lds2, stream>>>(h, w2, b2, dt, psi);

  curl_sample<<<(4 * HH * WW) / 256, 256, 0, stream>>>(psi, zr, zi, out);
}
